// SwinV2Block_55783035240626
// MI455X (gfx1250) — matrix operand order repaired and checked by reading, not compiled, not run on hardware
//
#include <hip/hip_runtime.h>
#include <hip/hip_bf16.h>
#include <math.h>

typedef __attribute__((ext_vector_type(16))) _Float16 v16h_t;
typedef __attribute__((ext_vector_type(8)))  _Float16 v8h_t;
typedef __attribute__((ext_vector_type(8)))  float    v8f_t;

#define WSZ   8
#define SSZ   4
#define NH    12
#define DIMC  384
#define HD    32
#define NTOK  64
#define NWIN  2048
#define HIDDEN 1536

// d_ws layout (bytes):
//   [0, 196608)            bias16  f32 [NH][64][64]
//   [196608, 1081344)      qkv_w   f16 [1152][384]
//   [1081344, 1376256)     proj_w  f16 [384][384]
//   [1376256, 2555904)     fc1_w   f16 [1536][384]
//   [2555904, 3735552)     fc2_w   f16 [384][1536]
#define WS_QKV  196608
#define WS_PROJ 1081344
#define WS_FC1  1376256
#define WS_FC2  2555904

__device__ __forceinline__ v8f_t wmma_f32_f16(v16h_t a, v16h_t b, v8f_t c) {
  return __builtin_amdgcn_wmma_f32_16x16x32_f16(false, a, false, b, (short)0, c, false, false);
}

// A-fragment (16x32 f16) from row-major LDS: lane = M, K contiguous (k = 16*(lane/16) + s)
__device__ __forceinline__ v16h_t load_a_lds(const _Float16* A, int lda, int row0, int kc, int lane) {
  int m = row0 + (lane & 15);
  int koff = (lane < 16) ? 0 : 16;
  const _Float16* p = A + m * lda + kc + koff;
  v8h_t lo = *(const v8h_t*)p;
  v8h_t hi = *(const v8h_t*)(p + 8);
  v16h_t r;
#pragma unroll
  for (int i = 0; i < 8; ++i) { r[i] = lo[i]; r[i + 8] = hi[i]; }
  return r;
}

// B-fragment where B[k][n] = S[n][k] (S row-major f16 in LDS, K contiguous)
__device__ __forceinline__ v16h_t load_b_ldsT(const _Float16* S, int lds_, int n0, int kc, int lane) {
  int n = lane & 15;
  int koff = (lane < 16) ? 0 : 16;
  const _Float16* p = S + (n0 + n) * lds_ + kc + koff;
  v8h_t lo = *(const v8h_t*)p;
  v8h_t hi = *(const v8h_t*)(p + 8);
  v16h_t r;
#pragma unroll
  for (int i = 0; i < 8; ++i) { r[i] = lo[i]; r[i + 8] = hi[i]; }
  return r;
}

// B-fragment where B[k][n] = W[n0+n][kbase+k] (W row-major f16 in GLOBAL)
__device__ __forceinline__ v16h_t load_b_wgt(const _Float16* W, int ldw, int n0, int kc, int lane) {
  int n = lane & 15;
  int koff = (lane < 16) ? 0 : 16;
  const v8h_t* p = (const v8h_t*)(W + (size_t)(n0 + n) * ldw + kc + koff);
  v8h_t lo = p[0];
  v8h_t hi = p[1];
  v16h_t r;
#pragma unroll
  for (int i = 0; i < 8; ++i) { r[i] = lo[i]; r[i + 8] = hi[i]; }
  return r;
}

// ---------------------------------------------------------------------------
// Kernel 0: f32 -> f16 weight conversion (runs once per launch, tiny)
// ---------------------------------------------------------------------------
__global__ __launch_bounds__(256) void k_cvt(const float* __restrict__ src,
                                             _Float16* __restrict__ dst, int n) {
  int i = (blockIdx.x * 256 + threadIdx.x) * 4;
  if (i < n) {
    float4 f = *(const float4*)(src + i);
    dst[i+0] = (_Float16)f.x; dst[i+1] = (_Float16)f.y;
    dst[i+2] = (_Float16)f.z; dst[i+3] = (_Float16)f.w;
  }
}

// ---------------------------------------------------------------------------
// Kernel 1: continuous position bias table -> 16*sigmoid(MLP(coords)) [NH,64,64]
// ---------------------------------------------------------------------------
__global__ __launch_bounds__(256) void k_bias(const float* __restrict__ w1,
                                              const float* __restrict__ b1,
                                              const float* __restrict__ w2,
                                              float* __restrict__ bias16) {
  __shared__ float hb[225][NH];
  int t = threadIdx.x;
  if (t < 225) {
    int rr = t / 15, cc = t % 15;
    float dy = (float)(rr - 7) * (8.0f / 7.0f);
    float dx = (float)(cc - 7) * (8.0f / 7.0f);
    float sy = (dy > 0.f) ? 1.f : ((dy < 0.f) ? -1.f : 0.f);
    float sx = (dx > 0.f) ? 1.f : ((dx < 0.f) ? -1.f : 0.f);
    float in0 = sy * log2f(fabsf(dy) + 1.f) * (1.0f / 3.0f);
    float in1 = sx * log2f(fabsf(dx) + 1.f) * (1.0f / 3.0f);
    float acc[NH];
#pragma unroll
    for (int h = 0; h < NH; ++h) acc[h] = 0.f;
    for (int j = 0; j < 512; ++j) {
      float hj = fmaxf(w1[2*j] * in0 + w1[2*j+1] * in1 + b1[j], 0.f);
#pragma unroll
      for (int h = 0; h < NH; ++h) acc[h] += hj * w2[h*512 + j];
    }
#pragma unroll
    for (int h = 0; h < NH; ++h) hb[t][h] = acc[h];
  }
  __syncthreads();
  for (int idx = t; idx < NH * 64 * 64; idx += 256) {
    int h = idx >> 12; int rem = idx & 4095;
    int q = rem >> 6, k = rem & 63;
    int dr = (q >> 3) - (k >> 3) + 7;
    int dc = (q & 7) - (k & 7) + 7;
    float b = hb[dr * 15 + dc][h];
    bias16[idx] = 16.f / (1.f + __expf(-b));
  }
}

// ---------------------------------------------------------------------------
// Kernel 2: fused shifted-window attention + projection + LN + residual -> y
// One workgroup (256 threads / 8 waves) per window.
// ---------------------------------------------------------------------------
__global__ __launch_bounds__(256) void k_attn(const float* __restrict__ x,
                                              const _Float16* __restrict__ qkv_w,
                                              const float* __restrict__ qkv_b,
                                              const float* __restrict__ lscale,
                                              const _Float16* __restrict__ proj_w,
                                              const float* __restrict__ proj_b,
                                              const float* __restrict__ n1g,
                                              const float* __restrict__ n1b,
                                              const float* __restrict__ bias16,
                                              float* __restrict__ yout) {
  __shared__ __attribute__((aligned(16))) char smem[98304 + 49152];
  __shared__ unsigned tokOff[NTOK];
  __shared__ int regid[NTOK];
  __shared__ float mu_s[NTOK], rs_s[NTOK];

  _Float16* Xs    = (_Float16*)smem;               // [64][384] f16
  float*    tmpf  = (float*)(smem + 49152);        // [64][32]  f32
  _Float16* Qs    = (_Float16*)(smem + 57344);     // [64][32]
  _Float16* Ks    = (_Float16*)(smem + 61440);     // [64][32]
  _Float16* Vt    = (_Float16*)(smem + 65536);     // [32][64] (transposed V)
  float*    attnf = (float*)(smem + 69632);        // [64][64] (bias async-preload + attn)
  _Float16* As    = (_Float16*)(smem + 86016);     // [64][64]
  float*    ypre  = (float*)smem;                  // [64][384] (aliases phase-1)
  _Float16* Os    = (_Float16*)(smem + 98304);     // [64][384]

  const int tid  = threadIdx.x;
  const int lane = tid & 31;
  const int wave = tid >> 5;

  const int w  = blockIdx.x;
  const int b  = w >> 6;
  const int wh = (w >> 3) & 7;
  const int ww = w & 7;

  if (tid < NTOK) {
    int i = tid >> 3, j = tid & 7;
    int gh = wh * WSZ + i, gw = ww * WSZ + j;
    int hs = (gh + SSZ) & 63, ws = (gw + SSZ) & 63;
    tokOff[tid] = ((unsigned)b * 4096u + (unsigned)(hs * 64 + ws)) * 384u;
    int rh = gh < 56 ? 0 : (gh < 60 ? 1 : 2);
    int rw = gw < 56 ? 0 : (gw < 60 ? 1 : 2);
    regid[tid] = rh * 3 + rw;
  }
  __syncthreads();

  // Stage shifted window X -> f16 LDS (4 threads per row)
  {
    int n = tid >> 2, part = tid & 3;
    const float4* src = (const float4*)(x + (size_t)tokOff[n] + part * 96);
    _Float16* dst = Xs + n * DIMC + part * 96;
#pragma unroll
    for (int i = 0; i < 24; ++i) {
      float4 f = src[i];
      dst[4*i+0] = (_Float16)f.x; dst[4*i+1] = (_Float16)f.y;
      dst[4*i+2] = (_Float16)f.z; dst[4*i+3] = (_Float16)f.w;
    }
  }
  __syncthreads();

  const unsigned attnf_lds = (unsigned)(size_t)(void*)attnf;

  for (int h = 0; h < NH; ++h) {
    // ---- async-preload this head's bias slab (16 KB) into attnf via the
    //      CDNA5 async LDS-DMA path; overlaps with the QKV WMMA phase ----
    {
      const float* gsrc = bias16 + h * 4096;
#pragma unroll
      for (int i = 0; i < 4; ++i) {
        unsigned e  = (unsigned)(tid + i * 256) * 16u;   // byte offset of 16B chunk
        unsigned lo = attnf_lds + e;
        asm volatile("global_load_async_to_lds_b128 %0, %1, %2"
                     :: "v"(lo), "v"(e), "s"(gsrc) : "memory");
      }
    }

    const float sc_h = __expf(fminf(lscale[h], 4.60517019f));

    // ---- Q, K, V projections: [64,384]@[384,32], one 16x16 tile per wave ----
    for (int p = 0; p < 3; ++p) {
      const int wrow0 = p * DIMC + h * HD;
      const int tm = wave >> 1, tn = wave & 1;
      v8f_t acc = {};
#pragma unroll 4
      for (int kc = 0; kc < DIMC; kc += 32) {
        v16h_t a  = load_a_lds(Xs, DIMC, tm * 16, kc, lane);
        v16h_t bb = load_b_wgt(qkv_w, DIMC, wrow0 + tn * 16, kc, lane);
        acc = wmma_f32_f16(a, bb, acc);
      }
      {
        float bias = qkv_b[wrow0 + tn * 16 + (lane & 15)];
        int nn = tn * 16 + (lane & 15);
        int mb = tm * 16 + ((lane >> 4) << 3);
#pragma unroll
        for (int r = 0; r < 8; ++r) tmpf[(mb + r) * HD + nn] = acc[r] + bias;
      }
      __syncthreads();
      if (tid < NTOK) {
        const float* row = tmpf + tid * HD;
        if (p < 2) {  // cosine-normalize q,k; fold logit scale into q
          float ss = 0.f;
#pragma unroll
          for (int d = 0; d < HD; ++d) ss += row[d] * row[d];
          float inv = 1.0f / fmaxf(sqrtf(ss), 1e-12f);
          if (p == 0) inv *= sc_h;
          _Float16* dst = (p == 0 ? Qs : Ks) + tid * HD;
#pragma unroll
          for (int d = 0; d < HD; ++d) dst[d] = (_Float16)(row[d] * inv);
        } else {      // V stored transposed [HD][NTOK]
#pragma unroll
          for (int d = 0; d < HD; ++d) Vt[d * NTOK + tid] = (_Float16)row[d];
        }
      }
      __syncthreads();
    }

    // all async bias chunks must be visible to every wave before the RMW below
    asm volatile("s_wait_asynccnt 0x0" ::: "memory");
    __syncthreads();

    // ---- attn = q^ k^ * scale + bias + mask : 16 tiles, 2 per wave ----
#pragma unroll
    for (int tt = 0; tt < 2; ++tt) {
      int t = wave * 2 + tt, tm = t >> 2, tn = t & 3;
      v8f_t acc = {};
      v16h_t a  = load_a_lds(Qs, HD, tm * 16, 0, lane);
      v16h_t bb = load_b_ldsT(Ks, HD, tn * 16, 0, lane);
      acc = wmma_f32_f16(a, bb, acc);
      int kcol = tn * 16 + (lane & 15);
      int mb = tm * 16 + ((lane >> 4) << 3);
      int rk = regid[kcol];
#pragma unroll
      for (int r = 0; r < 8; ++r) {
        int q = mb + r;
        float bv = attnf[q * 64 + kcol];   // async-preloaded bias
        attnf[q * 64 + kcol] = acc[r] + bv + ((regid[q] == rk) ? 0.f : -100.f);
      }
    }
    __syncthreads();

    // ---- softmax per query row ----
    if (tid < NTOK) {
      float* row = attnf + tid * 64;
      float mx = -3.4e38f;
#pragma unroll 8
      for (int j = 0; j < 64; ++j) mx = fmaxf(mx, row[j]);
      float s = 0.f;
#pragma unroll 8
      for (int j = 0; j < 64; ++j) { float e = __expf(row[j] - mx); row[j] = e; s += e; }
      float inv = 1.f / s;
      _Float16* arow = As + tid * 64;
#pragma unroll 8
      for (int j = 0; j < 64; ++j) arow[j] = (_Float16)(row[j] * inv);
    }
    __syncthreads();

    // ---- out_h = attn @ v -> Os[:, h*32 .. ] ----
    {
      int tm = wave >> 1, tn = wave & 1;
      v8f_t acc = {};
#pragma unroll
      for (int kc = 0; kc < 64; kc += 32) {
        v16h_t a  = load_a_lds(As, 64, tm * 16, kc, lane);
        v16h_t bb = load_b_ldsT(Vt, NTOK, tn * 16, kc, lane);
        acc = wmma_f32_f16(a, bb, acc);
      }
      int d = tn * 16 + (lane & 15);
      int mb = tm * 16 + ((lane >> 4) << 3);
#pragma unroll
      for (int r = 0; r < 8; ++r) Os[(mb + r) * DIMC + h * HD + d] = (_Float16)acc[r];
    }
    __syncthreads();
  }

  // ---- projection: ypre = Os @ proj_w^T + proj_b  (96 tiles, 12 per wave) ----
  for (int jt = 0; jt < 12; ++jt) {
    int tile = wave * 12 + jt;
    int tm = tile / 24, tn = tile % 24;
    v8f_t acc = {};
#pragma unroll 4
    for (int kc = 0; kc < DIMC; kc += 32) {
      v16h_t a  = load_a_lds(Os, DIMC, tm * 16, kc, lane);
      v16h_t bb = load_b_wgt(proj_w, DIMC, tn * 16, kc, lane);
      acc = wmma_f32_f16(a, bb, acc);
    }
    int nn = tn * 16 + (lane & 15);
    float pb = proj_b[nn];
    int mb = tm * 16 + ((lane >> 4) << 3);
#pragma unroll
    for (int r = 0; r < 8; ++r) ypre[(mb + r) * DIMC + nn] = acc[r] + pb;
  }
  __syncthreads();

  // ---- LayerNorm stats per token ----
  if (tid < NTOK) {
    const float* row = ypre + tid * DIMC;
    float s1 = 0.f, s2 = 0.f;
    for (int c = 0; c < DIMC; ++c) { float v = row[c]; s1 += v; s2 += v * v; }
    float m = s1 * (1.0f / DIMC);
    mu_s[tid] = m;
    rs_s[tid] = rsqrtf(fmaxf(s2 * (1.0f / DIMC) - m * m, 0.f) + 1e-5f);
  }
  __syncthreads();

  // ---- y = x + LN(ypre), scattered back through the reverse shift ----
  for (int idx = tid; idx < NTOK * DIMC; idx += 256) {
    int n = idx / DIMC, c = idx - n * DIMC;
    size_t g = (size_t)tokOff[n] + c;
    yout[g] = x[g] + (ypre[idx] - mu_s[n]) * rs_s[n] * n1g[c] + n1b[c];
  }
}

// ---------------------------------------------------------------------------
// Kernel 3: fused MLP (fc1+gelu+fc2) + LN + residual, row-wise per 64 tokens.
// fc1 processed in 12 chunks of 128 hidden units; fc2 accum in registers.
// ---------------------------------------------------------------------------
__global__ __launch_bounds__(256) void k_mlp(const _Float16* __restrict__ fc1_w,
                                             const float* __restrict__ fc1_b,
                                             const _Float16* __restrict__ fc2_w,
                                             const float* __restrict__ fc2_b,
                                             const float* __restrict__ n2g,
                                             const float* __restrict__ n2b,
                                             float* __restrict__ y) {
  __shared__ __attribute__((aligned(16))) char smem[98304];
  __shared__ float mu_s[64], rs_s[64];
  _Float16* Ys   = (_Float16*)smem;             // [64][384]
  _Float16* Hs   = (_Float16*)(smem + 49152);   // [64][128]
  float*    mlpf = (float*)smem;                // [64][384] (after GEMMs)

  const int tid  = threadIdx.x;
  const int lane = tid & 31;
  const int wave = tid >> 5;
  const size_t base = (size_t)blockIdx.x * 64 * DIMC;

  {
    int n = tid >> 2, part = tid & 3;
    const float4* src = (const float4*)(y + base + n * DIMC + part * 96);
    _Float16* dst = Ys + n * DIMC + part * 96;
#pragma unroll
    for (int i = 0; i < 24; ++i) {
      float4 f = src[i];
      dst[4*i+0] = (_Float16)f.x; dst[4*i+1] = (_Float16)f.y;
      dst[4*i+2] = (_Float16)f.z; dst[4*i+3] = (_Float16)f.w;
    }
  }
  __syncthreads();

  v8f_t zero = {};
  v8f_t accO[4][3];
#pragma unroll
  for (int tm = 0; tm < 4; ++tm)
#pragma unroll
    for (int tj = 0; tj < 3; ++tj) accO[tm][tj] = zero;

  for (int cc = 0; cc < 12; ++cc) {
    // fc1 chunk (N = 128): 32 tiles, 4 per wave -> gelu -> Hs
#pragma unroll
    for (int jt = 0; jt < 4; ++jt) {
      int tile = wave * 4 + jt;
      int tm = tile >> 3, tn8 = tile & 7;
      int nrow0 = cc * 128 + tn8 * 16;
      v8f_t acc = zero;
#pragma unroll 4
      for (int kc = 0; kc < DIMC; kc += 32) {
        v16h_t a  = load_a_lds(Ys, DIMC, tm * 16, kc, lane);
        v16h_t bb = load_b_wgt(fc1_w, DIMC, nrow0, kc, lane);
        acc = wmma_f32_f16(a, bb, acc);
      }
      float bias = fc1_b[nrow0 + (lane & 15)];
      int nn = tn8 * 16 + (lane & 15);
      int mb = tm * 16 + ((lane >> 4) << 3);
#pragma unroll
      for (int r = 0; r < 8; ++r) {
        float xv = acc[r] + bias;
        float gl = 0.5f * xv * (1.f + erff(xv * 0.70710678f));
        Hs[(mb + r) * 128 + nn] = (_Float16)gl;
      }
    }
    __syncthreads();
    // fc2 partial: each wave owns 12 output tiles, K-chunk = 128
#pragma unroll
    for (int tm = 0; tm < 4; ++tm) {
#pragma unroll
      for (int tj = 0; tj < 3; ++tj) {
        int tn = wave * 3 + tj;
        v8f_t acc = accO[tm][tj];
#pragma unroll
        for (int kc = 0; kc < 128; kc += 32) {
          v16h_t a  = load_a_lds(Hs, 128, tm * 16, kc, lane);
          v16h_t bb = load_b_wgt(fc2_w, HIDDEN, tn * 16, cc * 128 + kc, lane);
          acc = wmma_f32_f16(a, bb, acc);
        }
        accO[tm][tj] = acc;
      }
    }
    __syncthreads();
  }

  // write mlp rows (Ys/Hs are dead now; mlpf aliases them)
#pragma unroll
  for (int tm = 0; tm < 4; ++tm)
#pragma unroll
    for (int tj = 0; tj < 3; ++tj) {
      int tn = wave * 3 + tj;
      int nn = tn * 16 + (lane & 15);
      float bias = fc2_b[nn];
      int mb = tm * 16 + ((lane >> 4) << 3);
#pragma unroll
      for (int r = 0; r < 8; ++r) mlpf[(mb + r) * DIMC + nn] = accO[tm][tj][r] + bias;
    }
  __syncthreads();

  if (tid < 64) {
    const float* row = mlpf + tid * DIMC;
    float s1 = 0.f, s2 = 0.f;
    for (int c = 0; c < DIMC; ++c) { float v = row[c]; s1 += v; s2 += v * v; }
    float m = s1 * (1.0f / DIMC);
    mu_s[tid] = m;
    rs_s[tid] = rsqrtf(fmaxf(s2 * (1.0f / DIMC) - m * m, 0.f) + 1e-5f);
  }
  __syncthreads();

  for (int idx = tid; idx < 64 * DIMC; idx += 256) {
    int n = idx / DIMC, c = idx - n * DIMC;
    size_t g = base + (size_t)idx;
    y[g] = y[g] + (mlpf[idx] - mu_s[n]) * rs_s[n] * n2g[c] + n2b[c];
  }
}

extern "C" void kernel_launch(void* const* d_in, const int* in_sizes, int n_in,
                              void* d_out, int out_size, void* d_ws, size_t ws_size,
                              hipStream_t stream) {
  (void)in_sizes; (void)n_in; (void)out_size; (void)ws_size;
  const float* x      = (const float*)d_in[0];
  const float* qkv_w  = (const float*)d_in[3];
  const float* qkv_b  = (const float*)d_in[4];
  const float* lscale = (const float*)d_in[5];
  const float* cpb_w1 = (const float*)d_in[6];
  const float* cpb_b1 = (const float*)d_in[7];
  const float* cpb_w2 = (const float*)d_in[8];
  const float* proj_w = (const float*)d_in[9];
  const float* proj_b = (const float*)d_in[10];
  const float* n1g    = (const float*)d_in[11];
  const float* n1b    = (const float*)d_in[12];
  const float* n2g    = (const float*)d_in[13];
  const float* n2b    = (const float*)d_in[14];
  const float* fc1_w  = (const float*)d_in[15];
  const float* fc1_b  = (const float*)d_in[16];
  const float* fc2_w  = (const float*)d_in[17];
  const float* fc2_b  = (const float*)d_in[18];
  float* out = (float*)d_out;

  char* ws = (char*)d_ws;
  float*    bias16 = (float*)ws;
  _Float16* wq16   = (_Float16*)(ws + WS_QKV);
  _Float16* wp16   = (_Float16*)(ws + WS_PROJ);
  _Float16* wf1    = (_Float16*)(ws + WS_FC1);
  _Float16* wf2    = (_Float16*)(ws + WS_FC2);

  k_bias<<<dim3(1), dim3(256), 0, stream>>>(cpb_w1, cpb_b1, cpb_w2, bias16);
  k_cvt<<<dim3((1152*384/4 + 255)/256), dim3(256), 0, stream>>>(qkv_w, wq16, 1152*384);
  k_cvt<<<dim3((384*384/4 + 255)/256),  dim3(256), 0, stream>>>(proj_w, wp16, 384*384);
  k_cvt<<<dim3((1536*384/4 + 255)/256), dim3(256), 0, stream>>>(fc1_w, wf1, 1536*384);
  k_cvt<<<dim3((384*1536/4 + 255)/256), dim3(256), 0, stream>>>(fc2_w, wf2, 384*1536);

  k_attn<<<dim3(NWIN), dim3(256), 0, stream>>>(x, wq16, qkv_b, lscale, wp16, proj_b,
                                               n1g, n1b, bias16, out);
  k_mlp<<<dim3(NWIN), dim3(256), 0, stream>>>(wf1, fc1_b, wf2, fc2_b, n2g, n2b, out);
}
